// Connect_Cls_12575664243346
// MI455X (gfx1250) — matrix operand order repaired and checked by reading, not compiled, not run on hardware
//
#include <hip/hip_runtime.h>
#include <hip/hip_bf16.h>

// ---------------------------------------------------------------------------
// Problem constants (match reference)
// ---------------------------------------------------------------------------
#define NNODES   8192
#define F_IN     512
#define F2       1024      // 2*F_IN = K of GEMM1
#define F_MID    1024      // N of GEMM1
#define N_CLASS  2
#define NEDGE    65536
#define NROWS    131072    // 2*E
#define BN_EPS   1e-5f

typedef unsigned int       u32;
typedef unsigned short     u16;
typedef unsigned long long u64;
typedef __attribute__((ext_vector_type(16))) __bf16 v16bf;
typedef __attribute__((ext_vector_type(8)))  float  v8f;
typedef __attribute__((ext_vector_type(4)))  u32    u32x4;
typedef __attribute__((ext_vector_type(8)))  int    i32x8;
typedef __attribute__((ext_vector_type(4)))  int    i32x4;

// ---------------------------------------------------------------------------
// Helpers
// ---------------------------------------------------------------------------
__device__ __forceinline__ u16 f2bf(float f) {
  u32 u = __builtin_bit_cast(u32, f);
  u = u + 0x7FFFu + ((u >> 16) & 1u);   // round-to-nearest-even
  return (u16)(u >> 16);
}
__device__ __forceinline__ float bf2f(u16 h) {
  u32 u = ((u32)h) << 16;
  return __builtin_bit_cast(float, u);
}
// Build a 16-element bf16 fragment (8 VGPRs) from two 16-byte chunks.
__device__ __forceinline__ v16bf frag16(const u16* p0, const u16* p1) {
  union { u32x4 q[2]; v16bf v; } u;
  u.q[0] = *(const u32x4*)p0;   // elements 0..7
  u.q[1] = *(const u32x4*)p1;   // elements 8..15
  return u.v;
}

// ---------------------------------------------------------------------------
// K0a: input f32 -> bf16
// ---------------------------------------------------------------------------
__global__ __launch_bounds__(256) void k_cvt_input(const float* __restrict__ in,
                                                   u16* __restrict__ out) {
  int i = (blockIdx.x * 256 + threadIdx.x) * 4;
  if (i < NNODES * F_IN) {
    float4 v = *(const float4*)(in + i);
    ushort4 o;
    o.x = f2bf(v.x); o.y = f2bf(v.y); o.z = f2bf(v.z); o.w = f2bf(v.w);
    *(ushort4*)(out + i) = o;
  }
}

// ---------------------------------------------------------------------------
// K0b: W1 [K=1024][N=1024] f32 -> W1T [N][K] bf16  (LDS tile transpose)
// ---------------------------------------------------------------------------
__global__ __launch_bounds__(256) void k_transpose_w1(const float* __restrict__ W1,
                                                      u16* __restrict__ W1T) {
  __shared__ float tile[32][33];
  int kt = blockIdx.x * 32;          // K tile base
  int nt = blockIdx.y * 32;          // N tile base
  int tx = threadIdx.x & 31;
  int ty = threadIdx.x >> 5;         // 0..7
  #pragma unroll
  for (int j = ty; j < 32; j += 8)
    tile[j][tx] = W1[(kt + j) * F_MID + nt + tx];
  __syncthreads();
  #pragma unroll
  for (int j = ty; j < 32; j += 8)
    W1T[(nt + j) * F2 + kt + tx] = f2bf(tile[tx][j]);
}

// ---------------------------------------------------------------------------
// TDM: load a 128(N) x 64(K) bf16 tile of W1T into LDS.
//   D# per CDNA5 ISA ch.8: group0 = count/lds_addr/global_addr/type,
//   group1 = data_size, tensor dims/strides, tile dims, LDS padding.
//   Row data = 64 elems = 128B = 32 dwords; pad_interval code 4 (32 dwords)
//   inserts pad exactly once per row; pad_amount code 3 (+4 dwords = 16B)
//   => 144B LDS row stride (36 dwords).  Lane n row base dword 36n ==
//   4*(9n mod 16): 16 distinct 4-bank groups -> no 4-way ds_load_b128
//   conflicts; row base and +64B subtile base stay 16B aligned.
// ---------------------------------------------------------------------------
#define KSTEP    64
#define BROW_U16 72                        // padded LDS row stride in u16 (144B)
__device__ __forceinline__ void tdm_load_B(const u16* W1T, int bn, int kb,
                                           u16* lds_dst) {
  u64 ga = (u64)(size_t)(W1T + (size_t)bn * F2 + kb);
  u32 lds = (u32)(size_t)lds_dst;          // low 32 bits of flat addr = LDS offset
  u32x4 g0;
  g0[0] = 1u;                              // count=1 valid user descriptor
  g0[1] = lds;                             // lds_addr
  g0[2] = (u32)ga;                         // global_addr[31:0]
  g0[3] = (u32)(ga >> 32) | (2u << 30);    // global_addr[56:32] | type=2
  i32x8 g1;
  // data_size=1(2B) @16, pad_enable @20, pad_interval=4(32dw) @22, pad_amount=3(4dw) @25
  g1[0] = (int)((1u << 16) | (1u << 20) | (4u << 22) | (3u << 25));
  g1[1] = (int)((F2 & 0xFFFFu) << 16);                              // tensor_dim0 lo16 (atomic addr = 0)
  g1[2] = (int)(((F2 >> 16) & 0xFFFFu) | ((F_MID & 0xFFFFu) << 16));// dim0 hi16 | tensor_dim1 lo16
  g1[3] = (int)(((F_MID >> 16) & 0xFFFFu) | ((u32)KSTEP << 16));    // dim1 hi16 | tile_dim0=64
  g1[4] = 128;                                                      // tile_dim1=128 | tile_dim2=0
  g1[5] = F2;                                                       // tensor_dim0_stride lo32
  g1[6] = 0;                                                        // stride hi | dim1_stride lo
  g1[7] = 0;
  i32x4 gz4 = (i32x4)0;                    // groups 2/3 unused (2D tile)
  i32x8 gz8 = (i32x8)0;                    // extra descriptor group (unused)
  __builtin_amdgcn_tensor_load_to_lds(g0, g1, gz4, gz4, gz8, 0);
}

// ---------------------------------------------------------------------------
// K1: fused gather + GEMM1 (bf16 WMMA, f32 accum) + bias -> H bf16 [NROWS][1024]
//   Block: 256 threads = 8 waves; block tile 128(M) x 128(N)
//   Wave tile: 32(M) x 64(N) = 2x4 WMMA 16x16 accumulators
//   K stepped by 64 (2 x 32-K subtiles, 16 WMMA per barrier interval);
//   B tile staged in LDS by the Tensor Data Mover, double buffered.
// ---------------------------------------------------------------------------
__global__ __launch_bounds__(256) void k_gemm1(const u16* __restrict__ Abf,   // [NNODES][512]
                                               const u16* __restrict__ W1T,   // [1024][1024]
                                               const int* __restrict__ conn,  // [E][2]
                                               const int* __restrict__ disconn,
                                               const float* __restrict__ b1,
                                               u16* __restrict__ H) {
  __shared__ __align__(16) u16 smemB[2][128 * BROW_U16];   // 2 x 18 KB

  const int wave = threadIdx.x >> 5;
  const int lane = threadIdx.x & 31;
  const int l16  = lane & 15;
  const int hi   = lane >> 4;               // 0: low half-lanes, 1: high
  const int bm   = blockIdx.x * 128;
  const int bn   = blockIdx.y * 128;
  const int wm   = bm + (wave & 3) * 32;    // 4 waves along M
  const int wnl  = (wave >> 2) * 64;        // wave N offset inside block tile

  // Gather node indices for this lane's two M-subtile rows (uniform over K halves)
  int nodeA[2], nodeB[2];
  #pragma unroll
  for (int mt = 0; mt < 2; ++mt) {
    int r = wm + mt * 16 + l16;             // edge row 0..NROWS-1
    const int* idx = (r < NEDGE) ? (conn + 2 * r) : (disconn + 2 * (r - NEDGE));
    nodeA[mt] = idx[0];
    nodeB[mt] = idx[1];
  }

  v8f acc[2][4];
  #pragma unroll
  for (int mt = 0; mt < 2; ++mt)
    #pragma unroll
    for (int nt = 0; nt < 4; ++nt)
      acc[mt][nt] = (v8f)0.0f;

  // Prologue: kick off TDM for the first B tile.
  if (wave == 0) tdm_load_B(W1T, bn, 0, &smemB[0][0]);

  int ib = 0;
  for (int kb = 0; kb < F2; kb += KSTEP, ib ^= 1) {
    if (wave == 0) {
      if (kb + KSTEP < F2) {
        tdm_load_B(W1T, bn, kb + KSTEP, &smemB[ib ^ 1][0]);  // prefetch next tile
        __builtin_amdgcn_s_wait_tensorcnt(1);                // current tile landed
      } else {
        __builtin_amdgcn_s_wait_tensorcnt(0);
      }
    }
    __syncthreads();                                         // buf[ib] visible to all waves

    // 64-wide K chunk never crosses the 512 concat boundary: node select uniform.
    const bool first_half = (kb < F_IN);

    #pragma unroll
    for (int ks = 0; ks < 2; ++ks) {                         // two 32-K subtiles
      const int k0   = kb + ks * 32;
      const int colb = k0 & (F_IN - 1);                      // k0 % 512

      // A fragment: lanes<16: K 0..15 ; lanes>=16: K 16..31 (contiguous)
      v16bf af[2];
      #pragma unroll
      for (int mt = 0; mt < 2; ++mt) {
        int node = first_half ? nodeA[mt] : nodeB[mt];
        const u16* p = Abf + node * F_IN + colb + hi * 16;
        af[mt] = frag16(p, p + 8);
      }
      // B fragment from LDS: lanes<16: K 0..15 ; lanes>=16: K 16..31 (contiguous per row)
      v16bf bfr[4];
      #pragma unroll
      for (int nt = 0; nt < 4; ++nt) {
        const u16* p = &smemB[ib][(wnl + nt * 16 + l16) * BROW_U16 + ks * 32 + hi * 16];
        bfr[nt] = frag16(p, p + 8);
      }
      #pragma unroll
      for (int mt = 0; mt < 2; ++mt)
        #pragma unroll
        for (int nt = 0; nt < 4; ++nt)
          acc[mt][nt] = __builtin_amdgcn_wmma_f32_16x16x32_bf16(
              false, af[mt], false, bfr[nt], (short)0, acc[mt][nt], false, false);
    }

    __syncthreads();                        // all waves done with buf[ib] before reuse
  }

  // Epilogue: + bias, convert to bf16, store.
  // C layout: VGPR i holds M = i (+8 for lanes>=16), N = lane%16.
  #pragma unroll
  for (int nt = 0; nt < 4; ++nt) {
    int n = bn + wnl + nt * 16 + l16;
    float bias = b1[n];
    #pragma unroll
    for (int mt = 0; mt < 2; ++mt) {
      #pragma unroll
      for (int i = 0; i < 8; ++i) {
        int r = wm + mt * 16 + hi * 8 + i;
        H[r * F_MID + n] = f2bf(acc[mt][nt][i] + bias);
      }
    }
  }
}

// ---------------------------------------------------------------------------
// K2: deterministic column partial sums.
//   grid (rowgroups=256, colgroups=4); block 256 threads, 512 rows per group.
// ---------------------------------------------------------------------------
#define ROWS_PER_GROUP 512
#define N_ROWGROUPS    (NROWS / ROWS_PER_GROUP)   // 256
__global__ __launch_bounds__(256) void k_colstats(const u16* __restrict__ H,
                                                  float* __restrict__ partS,
                                                  float* __restrict__ partQ) {
  int rg  = blockIdx.x;                       // 0..255
  int col = blockIdx.y * 256 + threadIdx.x;   // 0..1023
  float s = 0.0f, q = 0.0f;
  int base = rg * ROWS_PER_GROUP;
  for (int r = 0; r < ROWS_PER_GROUP; ++r) {
    float v = bf2f(H[(base + r) * F_MID + col]);
    s += v;
    q += v * v;
  }
  partS[rg * F_MID + col] = s;
  partQ[rg * F_MID + col] = q;
}

// K2b: finalize per-column scale/shift: scale = gamma*rstd, shift = beta - mu*scale
__global__ __launch_bounds__(256) void k_finalize(const float* __restrict__ partS,
                                                  const float* __restrict__ partQ,
                                                  const float* __restrict__ gamma,
                                                  const float* __restrict__ beta,
                                                  float* __restrict__ scsh) {
  int col = blockIdx.x * 256 + threadIdx.x;
  float s = 0.0f, q = 0.0f;
  for (int g = 0; g < N_ROWGROUPS; ++g) {
    s += partS[g * F_MID + col];
    q += partQ[g * F_MID + col];
  }
  const float invN = 1.0f / (float)NROWS;
  float mu  = s * invN;
  float var = q * invN - mu * mu;
  float rstd = __frsqrt_rn(var + BN_EPS);
  float scale = gamma[col] * rstd;
  scsh[col]         = scale;
  scsh[F_MID + col] = beta[col] - mu * scale;
}

// ---------------------------------------------------------------------------
// K3: fused BN + ReLU + FC2 (N_CLASS=2). One row per wave32.
// ---------------------------------------------------------------------------
__global__ __launch_bounds__(256) void k_bn_relu_fc2(const u16* __restrict__ H,
                                                     const float* __restrict__ scsh,
                                                     const float* __restrict__ W2, // [1024][2]
                                                     const float* __restrict__ b2,
                                                     float* __restrict__ out) {
  int wave = threadIdx.x >> 5;
  int lane = threadIdx.x & 31;
  int row  = blockIdx.x * 8 + wave;
  const u16* hrow = H + row * F_MID;
  float a0 = 0.0f, a1 = 0.0f;
  #pragma unroll
  for (int i = 0; i < 8; ++i) {
    int c = i * 128 + lane * 4;               // 4 contiguous bf16 per lane
    uint2 pk = *(const uint2*)(hrow + c);
    u16 h0 = (u16)(pk.x & 0xFFFF), h1 = (u16)(pk.x >> 16);
    u16 h2 = (u16)(pk.y & 0xFFFF), h3 = (u16)(pk.y >> 16);
    #pragma unroll
    for (int j = 0; j < 4; ++j) {
      u16 hv = (j == 0) ? h0 : (j == 1) ? h1 : (j == 2) ? h2 : h3;
      int cc = c + j;
      float v = bf2f(hv) * scsh[cc] + scsh[F_MID + cc];
      v = fmaxf(v, 0.0f);
      a0 += v * W2[cc * 2 + 0];
      a1 += v * W2[cc * 2 + 1];
    }
  }
  #pragma unroll
  for (int off = 16; off >= 1; off >>= 1) {
    a0 += __shfl_xor(a0, off, 32);
    a1 += __shfl_xor(a1, off, 32);
  }
  if (lane == 0) {
    out[row * 2 + 0] = a0 + b2[0];
    out[row * 2 + 1] = a1 + b2[1];
  }
}

// ---------------------------------------------------------------------------
// Launch
// ---------------------------------------------------------------------------
extern "C" void kernel_launch(void* const* d_in, const int* in_sizes, int n_in,
                              void* d_out, int out_size, void* d_ws, size_t ws_size,
                              hipStream_t stream) {
  const float* input   = (const float*)d_in[0];
  const int*   conn    = (const int*)  d_in[1];
  const int*   disconn = (const int*)  d_in[2];
  const float* W1      = (const float*)d_in[3];
  const float* b1      = (const float*)d_in[4];
  const float* gamma   = (const float*)d_in[5];
  const float* beta    = (const float*)d_in[6];
  const float* W2      = (const float*)d_in[7];
  const float* b2      = (const float*)d_in[8];
  float* out = (float*)d_out;

  // Workspace carve-up (bytes)
  char* ws = (char*)d_ws;
  u16*   inBf  = (u16*)(ws);                                  // 8 MB
  u16*   W1T   = (u16*)(ws + 8388608);                        // 2 MB
  u16*   H     = (u16*)(ws + 10485760);                       // 256 MB
  float* partS = (float*)(ws + 278921216);                    // 1 MB
  float* partQ = (float*)(ws + 279969792);                    // 1 MB
  float* scsh  = (float*)(ws + 281018368);                    // 8 KB

  // K0a: input -> bf16
  k_cvt_input<<<dim3((NNODES * F_IN) / (256 * 4)), dim3(256), 0, stream>>>(input, inBf);
  // K0b: W1 -> W1T bf16
  k_transpose_w1<<<dim3(F2 / 32, F_MID / 32), dim3(256), 0, stream>>>(W1, W1T);
  // K1: GEMM1 (gather fused, TDM-staged B) -> H
  k_gemm1<<<dim3(NROWS / 128, F_MID / 128), dim3(256), 0, stream>>>(
      inBf, W1T, conn, disconn, b1, H);
  // K2: column partial stats
  k_colstats<<<dim3(N_ROWGROUPS, F_MID / 256), dim3(256), 0, stream>>>(H, partS, partQ);
  // K2b: finalize scale/shift
  k_finalize<<<dim3(F_MID / 256), dim3(256), 0, stream>>>(partS, partQ, gamma, beta, scsh);
  // K3: BN + ReLU + FC2
  k_bn_relu_fc2<<<dim3(NROWS / 8), dim3(256), 0, stream>>>(H, scsh, W2, b2, out);
}
